// NGP_49185965474600
// MI455X (gfx1250) — matrix operand order repaired and checked by reading, not compiled, not run on hardware
//
#include <hip/hip_runtime.h>

typedef __bf16 bf16_t;
typedef __attribute__((ext_vector_type(16))) __bf16 v16bf;
typedef __attribute__((ext_vector_type(8)))  __bf16 v8bf;
typedef __attribute__((ext_vector_type(8)))  float  v8f;

#define T_TBL 524288u

// ---- float -> bf16 via native hardware conversion (RNE on gfx1250)
__device__ __forceinline__ bf16_t f2bf(float f) {
  return (bf16_t)f;
}

// ---- A fragment (16x32 bf16) from row-major LDS tile.
// ISA layout: lanes 0-15 -> row (lane), K = 0..7 & 16..23; lanes 16-31 -> K = 8..15 & 24..31.
__device__ __forceinline__ v16bf load_a(const bf16_t* base, int stride, int row, int kbase) {
  const bf16_t* p = base + row * stride + kbase;
  v8bf lo = *(const v8bf*)(p);
  v8bf hi = *(const v8bf*)(p + 16);
  return __builtin_shufflevector(lo, hi, 0,1,2,3,4,5,6,7,8,9,10,11,12,13,14,15);
}

// ---- B fragment (32x16 bf16) from transposed weights WT[out][in] in LDS.
// ISA layout: lanes 0-15 -> col (lane), K = 0..7 & 16..23; lanes 16-31 -> K = 8..15 & 24..31.
__device__ __forceinline__ v16bf load_b(const bf16_t* wt, int stride, int n, int kbase) {
  const bf16_t* p = wt + n * stride + kbase;
  v8bf lo = *(const v8bf*)(p);
  v8bf hi = *(const v8bf*)(p + 16);
  return __builtin_shufflevector(lo, hi, 0,1,2,3,4,5,6,7,8,9,10,11,12,13,14,15);
}

template<int KSTEPS>
__device__ __forceinline__ v8f mma_tile(const bf16_t* A, int lda, int mtile,
                                        const bf16_t* WT, int ldw, int ntile, int lane) {
  v8f acc = {0.f,0.f,0.f,0.f,0.f,0.f,0.f,0.f};
  const int arow = mtile * 16 + (lane & 15);
  const int bn   = ntile * 16 + (lane & 15);
  const int ka   = (lane >> 4) * 8;
  const int kb   = (lane >> 4) * 8;
#pragma unroll
  for (int ks = 0; ks < KSTEPS; ++ks) {
    v16bf a = load_a(A, lda, arow, ks * 32 + ka);
    v16bf b = load_b(WT, ldw, bn,  ks * 32 + kb);
    acc = __builtin_amdgcn_wmma_f32_16x16x32_bf16(false, a, false, b, (short)0, acc,
                                                  false, false);
  }
  return acc;
}

// D layout: VGPR i holds row (lane/16)*8 + i (+16*mtile), col = lane%16.
__device__ __forceinline__ void store_relu(bf16_t* D, int ldd, int mtile, int ntile,
                                           v8f acc, int lane) {
  const int n = ntile * 16 + (lane & 15);
  const int rbase = mtile * 16 + (lane >> 4) * 8;
  union { v8f v; float f[8]; } u; u.v = acc;
#pragma unroll
  for (int i = 0; i < 8; ++i) {
    float r = u.f[i] > 0.f ? u.f[i] : 0.f;
    D[(rbase + i) * ldd + n] = f2bf(r);
  }
}

// Global fp32 W[Kin][Nact]  ->  LDS bf16 WT[n][k] (transposed, zero-padded)
__device__ __forceinline__ void load_wt(const float* __restrict__ W, bf16_t* dst,
                                        int Kin, int Nact, int Npad, int stride, int t) {
  const int tot = Npad * stride;
  for (int e = t; e < tot; e += 256) {
    int n = e / stride;
    int k = e - n * stride;
    float v = (k < Kin && n < Nact) ? W[k * Nact + n] : 0.f;
    dst[e] = f2bf(v);
  }
}

struct alignas(16) Smem {
  bf16_t wx0[64*104];   // xyz L0: 96(pad of 92) x 64, transposed, stride 104
  bf16_t wx1[64*72];    // xyz L1: 64 x 64
  bf16_t wx2[32*72];    // xyz L2: 64 x 32
  bf16_t wr0[64*72];    // rgb L0: 64(pad of 48) x 64
  bf16_t wr1[64*72];    // rgb L1
  bf16_t wr2[16*72];    // rgb L2: 64 x 16(pad of 3)
  bf16_t wb0[64*72];    // byp L0: 64(pad of 60) x 64
  bf16_t wb1[64*72];    // byp L1
  bf16_t wb2[16*72];    // byp L2
  bf16_t enc[128*104];  // [hash 0..31 | freq 32..91 | zero 92..95]
  bf16_t ha [128*72];
  bf16_t hb [128*72];
  bf16_t rgbin[128*72]; // [sh 0..15 | h 16..47 | zero 48..63]
  float  rgbacc[128*16];
};

__device__ const int c_res[16] = {16,20,25,32,40,50,64,80,101,128,161,203,256,322,406,512};

__global__ __launch_bounds__(256)
void ngp_forward(const float* __restrict__ x, const float* __restrict__ dirs,
                 const float* __restrict__ table,
                 const float* __restrict__ Wx0, const float* __restrict__ Wx1,
                 const float* __restrict__ Wx2,
                 const float* __restrict__ Wr0, const float* __restrict__ Wr1,
                 const float* __restrict__ Wr2,
                 const float* __restrict__ Wb0, const float* __restrict__ Wb1,
                 const float* __restrict__ Wb2,
                 float* __restrict__ out, int N)
{
  __shared__ Smem sm;
  const int t    = threadIdx.x;
  const int lane = t & 31;
  const int wave = t >> 5;
  const int tile0 = blockIdx.x * 128;

  // ---------------- phase 0: weights -> LDS ----------------
  load_wt(Wx0, sm.wx0, 92, 64, 64, 104, t);
  load_wt(Wx1, sm.wx1, 64, 64, 64,  72, t);
  load_wt(Wx2, sm.wx2, 64, 32, 32,  72, t);
  load_wt(Wr0, sm.wr0, 48, 64, 64,  72, t);
  load_wt(Wr1, sm.wr1, 64, 64, 64,  72, t);
  load_wt(Wr2, sm.wr2, 64,  3, 16,  72, t);
  load_wt(Wb0, sm.wb0, 60, 64, 64,  72, t);
  load_wt(Wb1, sm.wb1, 64, 64, 64,  72, t);
  load_wt(Wb2, sm.wb2, 64,  3, 16,  72, t);

  // ---------------- phase 1: encodings ----------------
  const int p    = t & 127;   // point in tile
  const int half = t >> 7;    // 2 threads per point (wave-uniform)
  const int gp   = tile0 + p;
  const float xn0 = (x[gp*3+0] + 1.f) * 0.5f;
  const float xn1 = (x[gp*3+1] + 1.f) * 0.5f;
  const float xn2 = (x[gp*3+2] + 1.f) * 0.5f;

  // hash-grid: this thread handles 8 of the 16 levels
  for (int j = 0; j < 8; ++j) {
    const int l = half * 8 + j;
    const int res = c_res[l];
    const float rf = (float)res;
    float px = xn0 * rf, py = xn1 * rf, pz = xn2 * rf;
    float fx = floorf(px), fy = floorf(py), fz = floorf(pz);
    float wx = px - fx, wy = py - fy, wz = pz - fz;
    unsigned ix = (unsigned)fx, iy = (unsigned)fy, iz = (unsigned)fz;
    const unsigned s = (unsigned)res + 1u;
    const bool dense = (s * s * s <= T_TBL);
    const float* tl = table + (size_t)l * (size_t)T_TBL * 2u;
    float f0 = 0.f, f1 = 0.f;
#pragma unroll
    for (int c = 0; c < 8; ++c) {
      unsigned cx = ix + ((c >> 2) & 1);
      unsigned cy = iy + ((c >> 1) & 1);
      unsigned cz = iz + (c & 1);
      float cw = ((c & 4) ? wx : 1.f - wx) *
                 ((c & 2) ? wy : 1.f - wy) *
                 ((c & 1) ? wz : 1.f - wz);
      unsigned idx;
      if (dense) idx = cx + cy * s + cz * s * s;
      else       idx = (cx * 1u ^ cy * 2654435761u ^ cz * 805459861u) & (T_TBL - 1u);
      const float2 f = *(const float2*)(tl + (size_t)idx * 2u);
      f0 += cw * f.x; f1 += cw * f.y;
    }
    sm.enc[p*104 + 2*l + 0] = f2bf(f0);
    sm.enc[p*104 + 2*l + 1] = f2bf(f1);
  }

  if (half == 0) {
    // frequency encoding: per dim [sin(pi*2^k*x) k=0..9 | cos ...]
    const float xns[3] = {xn0, xn1, xn2};
    for (int dim = 0; dim < 3; ++dim) {
      float v = 3.14159265358979323846f * xns[dim];
#pragma unroll
      for (int k = 0; k < 10; ++k) {
        sm.enc[p*104 + 32 + dim*20 + k]      = f2bf(__sinf(v));
        sm.enc[p*104 + 32 + dim*20 + 10 + k] = f2bf(__cosf(v));
        v *= 2.f;
      }
    }
    for (int k = 92; k < 96; ++k) sm.enc[p*104 + k] = f2bf(0.f);
    for (int k = 48; k < 64; ++k) sm.rgbin[p*72 + k] = f2bf(0.f);
  } else {
    // spherical harmonics (degree 4) of normalized direction
    float dx = dirs[gp*3+0], dy = dirs[gp*3+1], dz = dirs[gp*3+2];
    float inv = rsqrtf(dx*dx + dy*dy + dz*dz);
    float X = dx*inv, Y = dy*inv, Z = dz*inv;
    float x2 = X*X, y2 = Y*Y, z2 = Z*Z;
    float xy = X*Y, yz = Y*Z, xz = X*Z;
    float sh[16];
    sh[0]  = 0.28209479177387814f;
    sh[1]  = -0.48860251190291987f * Y;
    sh[2]  =  0.48860251190291987f * Z;
    sh[3]  = -0.48860251190291987f * X;
    sh[4]  =  1.0925484305920792f * xy;
    sh[5]  = -1.0925484305920792f * yz;
    sh[6]  =  0.94617469575756f * z2 - 0.31539156525252005f;
    sh[7]  = -1.0925484305920792f * xz;
    sh[8]  =  0.5462742152960396f * (x2 - y2);
    sh[9]  =  0.5900435899266435f * Y * (-3.f * x2 + y2);
    sh[10] =  2.890611442640554f * xy * Z;
    sh[11] =  0.4570457994644657f * Y * (1.f - 5.f * z2);
    sh[12] =  0.3731763325901154f * Z * (5.f * z2 - 3.f);
    sh[13] =  0.4570457994644657f * X * (1.f - 5.f * z2);
    sh[14] =  1.445305721320277f * Z * (x2 - y2);
    sh[15] =  0.5900435899266435f * X * (-x2 + 3.f * y2);
#pragma unroll
    for (int k = 0; k < 16; ++k) sm.rgbin[p*72 + k] = f2bf(sh[k]);
  }
  __syncthreads();

  // ---------------- xyz MLP ----------------
  for (int tile = wave; tile < 32; tile += 8) {           // [128x96] @ [96x64]
    int mt = tile >> 2, nt = tile & 3;
    v8f acc = mma_tile<3>(sm.enc, 104, mt, sm.wx0, 104, nt, lane);
    store_relu(sm.ha, 72, mt, nt, acc, lane);
  }
  __syncthreads();
  for (int tile = wave; tile < 32; tile += 8) {           // [128x64] @ [64x64]
    int mt = tile >> 2, nt = tile & 3;
    v8f acc = mma_tile<2>(sm.ha, 72, mt, sm.wx1, 72, nt, lane);
    store_relu(sm.hb, 72, mt, nt, acc, lane);
  }
  __syncthreads();
  for (int tile = wave; tile < 16; tile += 8) {           // [128x64] @ [64x32] -> h
    int mt = tile >> 1, nt = tile & 1;
    v8f acc = mma_tile<2>(sm.hb, 72, mt, sm.wx2, 72, nt, lane);
    const int n = nt * 16 + (lane & 15);
    const int rbase = mt * 16 + (lane >> 4) * 8;
    union { v8f v; float f[8]; } u; u.v = acc;
#pragma unroll
    for (int i = 0; i < 8; ++i)
      sm.rgbin[(rbase + i) * 72 + 16 + n] = f2bf(u.f[i]);  // no relu (last layer)
    if (nt == 0 && (lane & 15) == 0) {                     // sigma = exp(h[:,0])
#pragma unroll
      for (int i = 0; i < 8; ++i)
        out[tile0 + rbase + i] = __expf(u.f[i]);
    }
  }
  __syncthreads();

  // ---------------- rgb MLP ----------------
  for (int tile = wave; tile < 32; tile += 8) {           // [128x64] @ [64x64]
    int mt = tile >> 2, nt = tile & 3;
    v8f acc = mma_tile<2>(sm.rgbin, 72, mt, sm.wr0, 72, nt, lane);
    store_relu(sm.ha, 72, mt, nt, acc, lane);
  }
  __syncthreads();
  for (int tile = wave; tile < 32; tile += 8) {
    int mt = tile >> 2, nt = tile & 3;
    v8f acc = mma_tile<2>(sm.ha, 72, mt, sm.wr1, 72, nt, lane);
    store_relu(sm.hb, 72, mt, nt, acc, lane);
  }
  __syncthreads();
  for (int tile = wave; tile < 8; tile += 8) {            // [128x64] @ [64x16]
    v8f acc = mma_tile<2>(sm.hb, 72, tile, sm.wr2, 72, 0, lane);
    const int n = lane & 15;
    const int rbase = tile * 16 + (lane >> 4) * 8;
    union { v8f v; float f[8]; } u; u.v = acc;
#pragma unroll
    for (int i = 0; i < 8; ++i)
      sm.rgbacc[(rbase + i) * 16 + n] = u.f[i];
  }
  __syncthreads();

  // ---------------- bypass MLP (input = freq features, enc cols 32..95) ----------------
  for (int tile = wave; tile < 32; tile += 8) {
    int mt = tile >> 2, nt = tile & 3;
    v8f acc = mma_tile<2>(sm.enc + 32, 104, mt, sm.wb0, 72, nt, lane);
    store_relu(sm.ha, 72, mt, nt, acc, lane);
  }
  __syncthreads();
  for (int tile = wave; tile < 32; tile += 8) {
    int mt = tile >> 2, nt = tile & 3;
    v8f acc = mma_tile<2>(sm.ha, 72, mt, sm.wb1, 72, nt, lane);
    store_relu(sm.hb, 72, mt, nt, acc, lane);
  }
  __syncthreads();
  for (int tile = wave; tile < 8; tile += 8) {            // add + sigmoid + store rgb
    v8f acc = mma_tile<2>(sm.hb, 72, tile, sm.wb2, 72, 0, lane);
    const int n = lane & 15;
    const int rbase = tile * 16 + (lane >> 4) * 8;
    union { v8f v; float f[8]; } u; u.v = acc;
    if (n < 3) {
#pragma unroll
      for (int i = 0; i < 8; ++i) {
        float v = u.f[i] + sm.rgbacc[(rbase + i) * 16 + n];
        float sg = 1.f / (1.f + __expf(-v));
        out[N + (size_t)(tile0 + rbase + i) * 3 + n] = sg;
      }
    }
  }
}

extern "C" void kernel_launch(void* const* d_in, const int* in_sizes, int n_in,
                              void* d_out, int out_size, void* d_ws, size_t ws_size,
                              hipStream_t stream) {
  (void)n_in; (void)d_ws; (void)ws_size; (void)out_size;
  const float* x    = (const float*)d_in[0];
  const float* dirs = (const float*)d_in[1];
  const float* tbl  = (const float*)d_in[2];
  const float* Wx0  = (const float*)d_in[3];
  const float* Wx1  = (const float*)d_in[4];
  const float* Wx2  = (const float*)d_in[5];
  const float* Wr0  = (const float*)d_in[6];
  const float* Wr1  = (const float*)d_in[7];
  const float* Wr2  = (const float*)d_in[8];
  const float* Wb0  = (const float*)d_in[9];
  const float* Wb1  = (const float*)d_in[10];
  const float* Wb2  = (const float*)d_in[11];
  float* out = (float*)d_out;
  const int N = in_sizes[0] / 3;
  const int blocks = N / 128;
  ngp_forward<<<blocks, 256, 0, stream>>>(x, dirs, tbl,
                                          Wx0, Wx1, Wx2,
                                          Wr0, Wr1, Wr2,
                                          Wb0, Wb1, Wb2,
                                          out, N);
}
